// BertLayer_10548439679327
// MI455X (gfx1250) — matrix operand order repaired and checked by reading, not compiled, not run on hardware
//
#include <hip/hip_runtime.h>
#include <hip/hip_bf16.h>
#include <stdint.h>

typedef __attribute__((ext_vector_type(16))) __bf16 v16bf;
typedef __attribute__((ext_vector_type(8)))  __bf16 v8bf;
typedef __attribute__((ext_vector_type(8)))  float  v8f;

#define LP 72                      // LDS row pitch in bf16 elements (64 + 8 pad)
#define LOG2E 1.4426950408889634f

// ---------------------------------------------------------------------------
// Fragment loaders (layouts per CDNA5 ISA 7.12.2, wave32)
// ---------------------------------------------------------------------------
__device__ inline v16bf load_a_frag(const __bf16* lds, int mbase, int k0, int lane) {
    int m  = mbase + (lane & 15);
    int kb = (lane & 16) ? 8 : 0;
    const __bf16* p = lds + m * LP + k0 + kb;
    union { v16bf v; v8bf h[2]; } u;
    u.h[0] = *(const v8bf*)(p);        // K = k0+kb .. +7
    u.h[1] = *(const v8bf*)(p + 16);   // K = k0+kb+16 .. +23
    return u.v;
}

// lds is [n][k]; B-fragment K = k0+kb..+7 and k0+kb+16..+23 (kb = 0 or 8).
__device__ inline v16bf load_b_frag(const __bf16* lds, int nbase, int k0, int lane) {
    int n  = nbase + (lane & 15);
    int kb = (lane & 16) ? 8 : 0;
    const __bf16* p = lds + n * LP + k0 + kb;
    union { v16bf v; v8bf h[2]; } u;
    u.h[0] = *(const v8bf*)(p);
    u.h[1] = *(const v8bf*)(p + 16);
    return u.v;
}

// ---------------------------------------------------------------------------
// Async global->LDS copy of `rows` rows of 64 bf16 (128B) each, row-major,
// into LDS with pitch LP. Each lane copies 16B per instruction (ASYNCcnt).
// ---------------------------------------------------------------------------
__device__ inline void async_rows(const __bf16* g, long gstride, __bf16* lds,
                                  int tid, int nth, int rows) {
    uint32_t lbase = (uint32_t)(uintptr_t)lds;
    for (int c = tid; c < rows * 8; c += nth) {
        int row = c >> 3, ch = c & 7;
        uint64_t ga = (uint64_t)(uintptr_t)(g + (long)row * gstride + ch * 8);
        uint32_t la = lbase + (uint32_t)(row * (LP * 2) + ch * 16);
        asm volatile("global_load_async_to_lds_b128 %0, %1, off"
                     :: "v"(la), "v"(ga) : "memory");
    }
}

__device__ inline void wait_async_all() {
    asm volatile("s_wait_asynccnt 0x0" ::: "memory");
}

// ---------------------------------------------------------------------------
// fp32 -> bf16 conversion (plain and transposing variants)
// ---------------------------------------------------------------------------
__global__ __launch_bounds__(256) void cvt_bf16(const float* __restrict__ s,
                                                __bf16* __restrict__ d, long n) {
    long i = (long)blockIdx.x * blockDim.x + threadIdx.x;
    long st = (long)gridDim.x * blockDim.x;
    for (; i < n; i += st) d[i] = (__bf16)s[i];
}

// src [K,N] fp32 -> dst [N,K] bf16
__global__ __launch_bounds__(256) void cvt_bf16_t(const float* __restrict__ s,
                                                  __bf16* __restrict__ d, int K, int N) {
    long n = (long)K * N;
    long i = (long)blockIdx.x * blockDim.x + threadIdx.x;
    long st = (long)gridDim.x * blockDim.x;
    for (; i < n; i += st) {
        long k = i / N, c = i % N;
        d[c * (long)K + k] = (__bf16)s[i];
    }
}

// ---------------------------------------------------------------------------
// Generic bf16 GEMM:  C[M,N] = A[M,K] @ Bt[N,K]^T + bias, templated epilogue.
// 256 threads = 8 waves; block tile 128x128; wave tile 64x32; K-step 64.
// Double-buffered async global->LDS staging.
// EPI 0: +bias -> f32            EPI 1: +bias, exact GELU -> bf16
// EPI 2: +bias -> bf16 scattered to [B, NH, S, HD] head layout (Q/K)
// EPI 3: +bias -> bf16 scattered to [B, NH, HD, S] transposed layout (V)
// ---------------------------------------------------------------------------
template <int EPI>
__global__ __launch_bounds__(256) void gemm_bf16(
    const __bf16* __restrict__ A, const __bf16* __restrict__ Bt,
    const float* __restrict__ bias, void* __restrict__ out,
    int N, int K)
{
    __shared__ alignas(16) __bf16 lA[2][128 * LP];
    __shared__ alignas(16) __bf16 lB[2][128 * LP];

    const int tid  = threadIdx.x;
    const int lane = tid & 31;
    const int w    = tid >> 5;
    const int wm   = (w >> 2) * 64;   // 0 or 64
    const int wn   = (w & 3)  * 32;   // 0,32,64,96
    const long m0  = (long)blockIdx.y * 128;
    const long n0  = (long)blockIdx.x * 128;

    const __bf16* Abase = A  + m0 * K;
    const __bf16* Bbase = Bt + n0 * K;

    v8f acc[4][2] = {};

    const int T = K >> 6;
    async_rows(Abase, K, lA[0], tid, 256, 128);
    async_rows(Bbase, K, lB[0], tid, 256, 128);

    for (int t = 0; t < T; ++t) {
        const int cur = t & 1;
        wait_async_all();
        __syncthreads();
        if (t + 1 < T) {
            int k1 = (t + 1) << 6;
            async_rows(Abase + k1, K, lA[cur ^ 1], tid, 256, 128);
            async_rows(Bbase + k1, K, lB[cur ^ 1], tid, 256, 128);
        }
#pragma unroll
        for (int kc = 0; kc < 2; ++kc) {
            v16bf af[4], bfr[2];
#pragma unroll
            for (int i = 0; i < 4; ++i) af[i] = load_a_frag(lA[cur], wm + i * 16, kc * 32, lane);
#pragma unroll
            for (int j = 0; j < 2; ++j) bfr[j] = load_b_frag(lB[cur], wn + j * 16, kc * 32, lane);
#pragma unroll
            for (int i = 0; i < 4; ++i)
#pragma unroll
                for (int j = 0; j < 2; ++j)
                    acc[i][j] = __builtin_amdgcn_wmma_f32_16x16x32_bf16(
                        false, af[i], false, bfr[j], (short)0, acc[i][j], false, false);
        }
    }

    // Epilogue: C layout row = (lane&16 ? 8:0)+r, col = lane&15
    const int rbase = (lane & 16) ? 8 : 0;
    const int cl    = lane & 15;
#pragma unroll
    for (int i = 0; i < 4; ++i)
#pragma unroll
        for (int j = 0; j < 2; ++j)
#pragma unroll
            for (int r = 0; r < 8; ++r) {
                long row = m0 + wm + i * 16 + rbase + r;
                long col = n0 + wn + j * 16 + cl;
                float val = acc[i][j][r] + bias[col];
                if constexpr (EPI == 0) {
                    ((float*)out)[row * N + col] = val;
                } else if constexpr (EPI == 1) {
                    float g = 0.5f * val * (1.0f + erff(val * 0.70710678118654752f));
                    ((__bf16*)out)[row * N + col] = (__bf16)g;
                } else if constexpr (EPI == 2) {
                    long b = row >> 9, s = row & 511;        // S = 512
                    long h = col >> 6, d = col & 63;         // HD = 64
                    ((__bf16*)out)[(((b * 12 + h) * 512 + s) << 6) + d] = (__bf16)val;
                } else {                                      // V transposed: [bh][d][s]
                    long b = row >> 9, s = row & 511;
                    long h = col >> 6, d = col & 63;
                    ((__bf16*)out)[(((b * 12 + h) << 6) + d) * 512 + s] = (__bf16)val;
                }
            }
}

// ---------------------------------------------------------------------------
// Flash attention, one (head, 64-row Q tile) per 128-thread block (4 waves).
// q,k: bf16 [B*NH, 512, 64]; vt: bf16 [B*NH, 64, 512]; ctx: bf16 [M, 768].
// Double-buffered async K/V staging.
// ---------------------------------------------------------------------------
__global__ __launch_bounds__(128) void attn_kernel(
    const __bf16* __restrict__ q, const __bf16* __restrict__ k,
    const __bf16* __restrict__ vt, __bf16* __restrict__ ctx)
{
    __shared__ alignas(16) __bf16 lq[64 * LP];
    __shared__ alignas(16) __bf16 lk[2][64 * LP];
    __shared__ alignas(16) __bf16 lvt[2][64 * LP];
    __shared__ alignas(16) __bf16 lp[64 * LP];

    const int tid = threadIdx.x, lane = tid & 31, w = tid >> 5;
    const int bh = blockIdx.x >> 3;   // 0..191
    const int qt = blockIdx.x & 7;    // q tile
    const long base  = (long)bh * 512 * 64;   // q/k head base
    const long vbase = (long)bh * 64 * 512;   // vt head base
    const int rbase = (lane & 16) ? 8 : 0;
    const int cl    = lane & 15;

    // Stage Q tile (64x64) + first K/V tiles
    async_rows(q + base + (long)qt * 4096, 64, lq, tid, 128, 64);
    async_rows(k + base, 64, lk[0], tid, 128, 64);
    async_rows(vt + vbase, 512, lvt[0], tid, 128, 64);

    v16bf qa0, qa1;
    v8f o[4] = {};
    float mrow[8], lrow[8];
#pragma unroll
    for (int r = 0; r < 8; ++r) { mrow[r] = -3.0e38f; lrow[r] = 0.f; }

    for (int kt = 0; kt < 8; ++kt) {
        const int cur = kt & 1;
        wait_async_all();
        __syncthreads();
        if (kt == 0) {
            qa0 = load_a_frag(lq, w * 16, 0, lane);
            qa1 = load_a_frag(lq, w * 16, 32, lane);
        }
        if (kt + 1 < 8) {
            async_rows(k + base + (long)(kt + 1) * 4096, 64, lk[cur ^ 1], tid, 128, 64);
            async_rows(vt + vbase + (long)(kt + 1) * 64, 512, lvt[cur ^ 1], tid, 128, 64);
        }

        // S = (Q K^T) * 1/8 ; K rows are B-matrix columns (contiguous d)
        v8f sj[4];
        float rmax[8];
#pragma unroll
        for (int r = 0; r < 8; ++r) rmax[r] = -3.0e38f;
#pragma unroll
        for (int j = 0; j < 4; ++j) {
            v16bf kb0 = load_b_frag(lk[cur], j * 16, 0, lane);
            v16bf kb1 = load_b_frag(lk[cur], j * 16, 32, lane);
            v8f s = {};
            s = __builtin_amdgcn_wmma_f32_16x16x32_bf16(false, qa0, false, kb0, (short)0, s, false, false);
            s = __builtin_amdgcn_wmma_f32_16x16x32_bf16(false, qa1, false, kb1, (short)0, s, false, false);
#pragma unroll
            for (int r = 0; r < 8; ++r) { s[r] *= 0.125f; rmax[r] = fmaxf(rmax[r], s[r]); }
            sj[j] = s;
        }
        // row max over 16 lanes (columns)
#pragma unroll
        for (int off = 8; off; off >>= 1)
#pragma unroll
            for (int r = 0; r < 8; ++r) rmax[r] = fmaxf(rmax[r], __shfl_xor(rmax[r], off, 32));

        float mnew[8], alpha[8], rsum[8], ps[4][8];
#pragma unroll
        for (int r = 0; r < 8; ++r) {
            mnew[r]  = fmaxf(mrow[r], rmax[r]);
            alpha[r] = exp2f((mrow[r] - mnew[r]) * LOG2E);
            rsum[r]  = 0.f;
        }
#pragma unroll
        for (int j = 0; j < 4; ++j)
#pragma unroll
            for (int r = 0; r < 8; ++r) {
                float pv = exp2f((sj[j][r] - mnew[r]) * LOG2E);
                ps[j][r] = pv; rsum[r] += pv;
            }
#pragma unroll
        for (int off = 8; off; off >>= 1)
#pragma unroll
            for (int r = 0; r < 8; ++r) rsum[r] += __shfl_xor(rsum[r], off, 32);
#pragma unroll
        for (int r = 0; r < 8; ++r) {
            lrow[r] = lrow[r] * alpha[r] + rsum[r];
            mrow[r] = mnew[r];
        }
#pragma unroll
        for (int j = 0; j < 4; ++j)
#pragma unroll
            for (int r = 0; r < 8; ++r) o[j][r] *= alpha[r];

        // P -> LDS (bf16) to re-enter A-fragment layout
#pragma unroll
        for (int j = 0; j < 4; ++j)
#pragma unroll
            for (int r = 0; r < 8; ++r)
                lp[(w * 16 + rbase + r) * LP + j * 16 + cl] = (__bf16)ps[j][r];
        __syncthreads();

        v16bf pa0 = load_a_frag(lp, w * 16, 0, lane);
        v16bf pa1 = load_a_frag(lp, w * 16, 32, lane);
#pragma unroll
        for (int j = 0; j < 4; ++j) {
            v16bf vb0 = load_b_frag(lvt[cur], j * 16, 0, lane);
            v16bf vb1 = load_b_frag(lvt[cur], j * 16, 32, lane);
            o[j] = __builtin_amdgcn_wmma_f32_16x16x32_bf16(false, pa0, false, vb0, (short)0, o[j], false, false);
            o[j] = __builtin_amdgcn_wmma_f32_16x16x32_bf16(false, pa1, false, vb1, (short)0, o[j], false, false);
        }
    }

    // Normalize and scatter to ctx [token, head*64 + d]
    const int b = bh / 12, head = bh % 12;
#pragma unroll
    for (int j = 0; j < 4; ++j)
#pragma unroll
        for (int r = 0; r < 8; ++r) {
            int qrow = qt * 64 + w * 16 + rbase + r;
            long tok = (long)b * 512 + qrow;
            float val = o[j][r] / lrow[r];
            ctx[tok * 768 + head * 64 + j * 16 + cl] = (__bf16)val;
        }
}

// ---------------------------------------------------------------------------
// out = LayerNorm(a + b) * g + beta ; fp32 out (+ optional bf16 copy)
// one 256-thread block per row of 768
// ---------------------------------------------------------------------------
__global__ __launch_bounds__(256) void add_ln(
    const float* __restrict__ a, const float* __restrict__ bb,
    const float* __restrict__ g, const float* __restrict__ beta,
    float* __restrict__ outf, __bf16* __restrict__ outb)
{
    __shared__ float red[256];
    const long row = blockIdx.x;
    const int tid = threadIdx.x;
    const float* pa = a  + row * 768;
    const float* pb = bb + row * 768;

    float v[3], s = 0.f;
#pragma unroll
    for (int i = 0; i < 3; ++i) { v[i] = pa[tid + i * 256] + pb[tid + i * 256]; s += v[i]; }
    red[tid] = s; __syncthreads();
    for (int o = 128; o; o >>= 1) { if (tid < o) red[tid] += red[tid + o]; __syncthreads(); }
    float mu = red[0] * (1.0f / 768.0f);
    __syncthreads();

    float s2 = 0.f;
#pragma unroll
    for (int i = 0; i < 3; ++i) { float d = v[i] - mu; s2 += d * d; }
    red[tid] = s2; __syncthreads();
    for (int o = 128; o; o >>= 1) { if (tid < o) red[tid] += red[tid + o]; __syncthreads(); }
    float rstd = rsqrtf(red[0] * (1.0f / 768.0f) + 1e-12f);

#pragma unroll
    for (int i = 0; i < 3; ++i) {
        int col = tid + i * 256;
        float y = (v[i] - mu) * rstd * g[col] + beta[col];
        outf[row * 768 + col] = y;
        if (outb) outb[row * 768 + col] = (__bf16)y;
    }
}

// ---------------------------------------------------------------------------
extern "C" void kernel_launch(void* const* d_in, const int* in_sizes, int n_in,
                              void* d_out, int out_size, void* d_ws, size_t ws_size,
                              hipStream_t stream) {
    (void)in_sizes; (void)n_in; (void)out_size; (void)ws_size;
    const long M = 8192, H = 768, FF = 3072;

    const float* x    = (const float*)d_in[0];
    const float* Wq   = (const float*)d_in[1];
    const float* bq   = (const float*)d_in[2];
    const float* Wk   = (const float*)d_in[3];
    const float* bk   = (const float*)d_in[4];
    const float* Wv   = (const float*)d_in[5];
    const float* bv   = (const float*)d_in[6];
    const float* Wo   = (const float*)d_in[7];
    const float* bo   = (const float*)d_in[8];
    const float* ln1g = (const float*)d_in[9];
    const float* ln1b = (const float*)d_in[10];
    const float* W1   = (const float*)d_in[11];
    const float* b1   = (const float*)d_in[12];
    const float* W2   = (const float*)d_in[13];
    const float* b2   = (const float*)d_in[14];
    const float* ln2g = (const float*)d_in[15];
    const float* ln2b = (const float*)d_in[16];

    char* ws = (char*)d_ws;
    size_t off = 0;
    auto alloc = [&](size_t bytes) { char* p = ws + off; off += (bytes + 255) & ~(size_t)255; return p; };

    __bf16* xb   = (__bf16*)alloc(M * H * 2);
    __bf16* qh   = (__bf16*)alloc(M * H * 2);
    __bf16* kh   = (__bf16*)alloc(M * H * 2);
    __bf16* vth  = (__bf16*)alloc(M * H * 2);   // V transposed per head [bh][d][s]
    __bf16* ctx  = (__bf16*)alloc(M * H * 2);
    __bf16* hb   = (__bf16*)alloc(M * H * 2);
    float*  tmpf = (float*)alloc(M * H * 4);    // attn_out, reused for ffn2 out
    float*  hf   = (float*)alloc(M * H * 4);
    __bf16* f1b  = (__bf16*)alloc(M * FF * 2);
    __bf16* Wqt  = (__bf16*)alloc(H * H * 2);   // all weights stored transposed [N][K]
    __bf16* Wkt  = (__bf16*)alloc(H * H * 2);
    __bf16* Wvt  = (__bf16*)alloc(H * H * 2);
    __bf16* Wot  = (__bf16*)alloc(H * H * 2);
    __bf16* W1t  = (__bf16*)alloc(H * FF * 2);
    __bf16* W2t  = (__bf16*)alloc(FF * H * 2);

    // fp32 -> bf16 (weights transposed to [N][K]) once
    cvt_bf16  <<<2048, 256, 0, stream>>>(x,  xb, M * H);
    cvt_bf16_t<<<1024, 256, 0, stream>>>(Wq, Wqt, (int)H, (int)H);
    cvt_bf16_t<<<1024, 256, 0, stream>>>(Wk, Wkt, (int)H, (int)H);
    cvt_bf16_t<<<1024, 256, 0, stream>>>(Wv, Wvt, (int)H, (int)H);
    cvt_bf16_t<<<1024, 256, 0, stream>>>(Wo, Wot, (int)H, (int)H);
    cvt_bf16_t<<<2048, 256, 0, stream>>>(W1, W1t, (int)H, (int)FF);
    cvt_bf16_t<<<2048, 256, 0, stream>>>(W2, W2t, (int)FF, (int)H);

    dim3 gridH(H / 128, M / 128);    // (6, 64)
    dim3 gridF(FF / 128, M / 128);   // (24, 64)

    // QKV projections -> head layouts (V transposed)
    gemm_bf16<2><<<gridH, 256, 0, stream>>>(xb, Wqt, bq, qh,  (int)H, (int)H);
    gemm_bf16<2><<<gridH, 256, 0, stream>>>(xb, Wkt, bk, kh,  (int)H, (int)H);
    gemm_bf16<3><<<gridH, 256, 0, stream>>>(xb, Wvt, bv, vth, (int)H, (int)H);

    // Flash attention: 192 heads * 8 q-tiles
    attn_kernel<<<1536, 128, 0, stream>>>(qh, kh, vth, ctx);

    // Output projection -> f32
    gemm_bf16<0><<<gridH, 256, 0, stream>>>(ctx, Wot, bo, tmpf, (int)H, (int)H);

    // h = LN(attn_out + x)
    add_ln<<<M, 256, 0, stream>>>(tmpf, x, ln1g, ln1b, hf, hb);

    // FFN1 + GELU -> bf16
    gemm_bf16<1><<<gridF, 256, 0, stream>>>(hb, W1t, b1, f1b, (int)FF, (int)H);

    // FFN2 -> f32 (reuse tmpf)
    gemm_bf16<0><<<gridH, 256, 0, stream>>>(f1b, W2t, b2, tmpf, (int)H, (int)FF);

    // out = LN(ffn + h)
    add_ln<<<M, 256, 0, stream>>>(tmpf, hf, ln2g, ln2b, (float*)d_out, nullptr);
}
